// GroupedQueryAttention_46093589020797
// MI455X (gfx1250) — matrix operand order repaired and checked by reading, not compiled, not run on hardware
//
#include <hip/hip_runtime.h>
#include <hip/hip_bf16.h>

#define B_   2
#define T_   2048
#define D_   1024
#define H_   16
#define KV_  4
#define DH_  64

typedef __attribute__((ext_vector_type(16))) _Float16 v16h;
typedef __attribute__((ext_vector_type(8)))  _Float16 v8h;
typedef __attribute__((ext_vector_type(4)))  _Float16 v4h;
typedef __attribute__((ext_vector_type(8)))  float    v8f;
typedef __attribute__((ext_vector_type(4)))  float    v4f;

static __device__ __forceinline__ v16h cat8(v8h lo, v8h hi) {
  return __builtin_shufflevector(lo, hi, 0,1,2,3,4,5,6,7,8,9,10,11,12,13,14,15);
}

// A tile fragment: lane pointer p = row(lane&15) base + (lane>>4)*8 halves.
// K chunks {c0..c0+7} and {16+c0..16+c0+7} per ISA 16-bit A layout.
static __device__ __forceinline__ v16h frag_a(const _Float16* p) {
  return cat8(*(const v8h*)p, *(const v8h*)(p + 16));
}
// B tile fragment: lane pointer p = col(lane&15) base + (lane>>4)*8 halves,
// K chunks {c0..c0+7} and {16+c0..16+c0+7}, matching the A layout.
static __device__ __forceinline__ v16h frag_b(const _Float16* p) {
  return cat8(*(const v8h*)p, *(const v8h*)(p + 16));
}

static __device__ __forceinline__ v8f wmma16(v16h a, v16h b, v8f c) {
  return __builtin_amdgcn_wmma_f32_16x16x32_f16(false, a, false, b, (short)0, c, false, false);
}

// -------------------------------------------------------------------------
// Kernel 0: fp32 -> f16 elementwise conversion (x and weights), vectorized x4.
// -------------------------------------------------------------------------
__global__ void __launch_bounds__(256) cvt_f32_f16_kernel(
    const float* __restrict__ in, _Float16* __restrict__ out, int n4) {
  const int i = blockIdx.x * 256 + threadIdx.x;
  if (i < n4) {
    v4f v = *(const v4f*)(in + (size_t)i * 4);
    v4h h;
#pragma unroll
    for (int j = 0; j < 4; ++j) h[j] = (_Float16)v[j];
    *(v4h*)(out + (size_t)i * 4) = h;
  }
}

// -------------------------------------------------------------------------
// Kernel 1: fused QKV projection + RoPE.  One wave = 32(row) x 64(col) tile
// of the [B*T, 1536] (q|k|v) GEMM.  8 WMMA per 12 b128 loads per K-step.
// -------------------------------------------------------------------------
__global__ void __launch_bounds__(256) qkv_rope_kernel(
    const _Float16* __restrict__ xh, const float* __restrict__ rope,
    const _Float16* __restrict__ wq, const _Float16* __restrict__ wk,
    const _Float16* __restrict__ wv,
    _Float16* __restrict__ qh, _Float16* __restrict__ kh,
    _Float16* __restrict__ vT) {
  const int lane = threadIdx.x & 31;
  const int wave = threadIdx.x >> 5;
  const int tile = blockIdx.x * 8 + wave;
  const int NCT = (D_ + 2 * KV_ * DH_) / 64;     // 24 column tiles
  const int m0 = (tile / NCT) * 32;
  const int c0 = (tile % NCT) * 64;

  const _Float16* W; int buf, rowsel;
  if (c0 < D_)            { W = wq; buf = 0; rowsel = c0; }
  else if (c0 < D_ + 256) { W = wk; buf = 1; rowsel = c0 - D_; }
  else                    { W = wv; buf = 2; rowsel = c0 - D_ - 256; }

  const _Float16* ap0 = xh + ((size_t)m0 + (lane & 15)) * D_ + ((lane >> 4) << 3);
  const _Float16* ap1 = ap0 + (size_t)16 * D_;
  const _Float16* bp0 = W + ((size_t)(rowsel +  0) + (lane & 15)) * D_ + ((lane >> 4) << 3);
  const _Float16* bp1 = bp0 + (size_t)16 * D_;
  const _Float16* bp2 = bp0 + (size_t)32 * D_;
  const _Float16* bp3 = bp0 + (size_t)48 * D_;

  v8f acc[2][4] = {{v8f{}, v8f{}, v8f{}, v8f{}}, {v8f{}, v8f{}, v8f{}, v8f{}}};
#pragma unroll 4
  for (int kk = 0; kk < D_; kk += 32) {
    v16h a0 = frag_a(ap0 + kk);
    v16h a1 = frag_a(ap1 + kk);
    v16h b0 = frag_b(bp0 + kk);
    v16h b1 = frag_b(bp1 + kk);
    v16h b2 = frag_b(bp2 + kk);
    v16h b3 = frag_b(bp3 + kk);
    acc[0][0] = wmma16(a0, b0, acc[0][0]);
    acc[1][0] = wmma16(a1, b0, acc[1][0]);
    acc[0][1] = wmma16(a0, b1, acc[0][1]);
    acc[1][1] = wmma16(a1, b1, acc[1][1]);
    acc[0][2] = wmma16(a0, b2, acc[0][2]);
    acc[1][2] = wmma16(a1, b2, acc[1][2]);
    acc[0][3] = wmma16(a0, b3, acc[0][3]);
    acc[1][3] = wmma16(a1, b3, acc[1][3]);
  }

  const int rbase = (lane >> 4) << 3;
  const int ncol  = lane & 15;
  const int b     = m0 / T_;                     // 32-row tile never crosses batch
#pragma unroll
  for (int mh = 0; mh < 2; ++mh) {
#pragma unroll
    for (int nt = 0; nt < 4; ++nt) {
      const int dh = ncol + nt * 16;             // 0..63 within head
#pragma unroll
      for (int r = 0; r < 8; ++r) {
        float v = acc[mh][nt][r];
        const int t = (m0 + 16 * mh + rbase + r) & (T_ - 1);
        float outv = v;
        if (buf != 2) {                          // RoPE for q and k
          float other = __shfl_xor(v, 1);        // (re,im) partner lane
          const int j = dh >> 1;
          const float c = rope[(t * (DH_ / 2) + j) * 2 + 0];
          const float s = rope[(t * (DH_ / 2) + j) * 2 + 1];
          outv = (dh & 1) ? (other * s + v * c) : (v * c - other * s);
        }
        const _Float16 hv = (_Float16)outv;
        if (buf == 0) {
          qh[(((size_t)b * H_ + (c0 >> 6)) * T_ + t) * DH_ + dh] = hv;
        } else if (buf == 1) {
          kh[(((size_t)b * KV_ + (rowsel >> 6)) * T_ + t) * DH_ + dh] = hv;
        } else {
          vT[(((size_t)b * KV_ + (rowsel >> 6)) * DH_ + dh) * T_ + t] = hv;
        }
      }
    }
  }
}

// -------------------------------------------------------------------------
// Kernel 2: flash attention.  One wave = (b, h, 32-query-row tile).
// 16 WMMA per 32-key chunk (8 for S, 8 for O) against 16 b128 K/V loads.
// -------------------------------------------------------------------------
__global__ void __launch_bounds__(256) attn_kernel(
    const _Float16* __restrict__ qh, const _Float16* __restrict__ kh,
    const _Float16* __restrict__ vT, _Float16* __restrict__ oh) {
  __shared__ _Float16 plds[8][32 * 32];
  const int lane = threadIdx.x & 31;
  const int wave = threadIdx.x >> 5;
  const int wid  = blockIdx.x * 8 + wave;
  const int qt = wid & (T_ / 32 - 1);            // 64 q-tiles
  const int h  = (wid >> 6) & (H_ - 1);
  const int b  = wid >> 10;
  const int q0 = qt * 32;
  const int kvh = h >> 2;                        // GQA mapping

  const _Float16* Q = qh + (((size_t)b * H_ + h) * T_ + q0) * DH_;
  const _Float16* K = kh + ((size_t)b * KV_ + kvh) * T_ * DH_;
  const _Float16* V = vT + ((size_t)b * KV_ + kvh) * DH_ * T_;

  const _Float16* qp = Q + (size_t)(lane & 15) * DH_ + ((lane >> 4) << 3);
  const _Float16* kp = K + (size_t)(lane & 15) * DH_ + ((lane >> 4) << 3);
  const _Float16* vp = V + (size_t)(lane & 15) * T_  + ((lane >> 4) << 3);
  _Float16* pp = &plds[wave][(lane & 15) * 32 + ((lane >> 4) << 3)];

  v16h aQ[2][2];
#pragma unroll
  for (int mt = 0; mt < 2; ++mt)
#pragma unroll
    for (int c = 0; c < 2; ++c)
      aQ[mt][c] = frag_a(qp + (size_t)(mt * 16) * DH_ + c * 32);

  v8f o[2][4] = {{v8f{}, v8f{}, v8f{}, v8f{}}, {v8f{}, v8f{}, v8f{}, v8f{}}};
  float mrow[2][8], lrow[2][8];
#pragma unroll
  for (int mt = 0; mt < 2; ++mt)
#pragma unroll
    for (int r = 0; r < 8; ++r) { mrow[mt][r] = -1e30f; lrow[mt][r] = 0.f; }

  const int rbase = (lane >> 4) << 3;
  const int ncol  = lane & 15;
  const float scale = 0.125f;                    // 1/sqrt(64)

  for (int kt = 0; kt < q0 + 32; kt += 32) {
    __builtin_prefetch(K + (size_t)(kt + 32) * DH_, 0, 1);
    // ---- K B-tiles: [nt = key half][c = dh chunk], shared by both M tiles ----
    v16h bk00 = frag_b(kp + (size_t)kt * DH_);
    v16h bk01 = frag_b(kp + (size_t)kt * DH_ + 32);
    v16h bk10 = frag_b(kp + (size_t)(kt + 16) * DH_);
    v16h bk11 = frag_b(kp + (size_t)(kt + 16) * DH_ + 32);
    v8f s[2][2] = {{v8f{}, v8f{}}, {v8f{}, v8f{}}};
    s[0][0] = wmma16(aQ[0][0], bk00, s[0][0]);
    s[1][0] = wmma16(aQ[1][0], bk00, s[1][0]);
    s[0][0] = wmma16(aQ[0][1], bk01, s[0][0]);
    s[1][0] = wmma16(aQ[1][1], bk01, s[1][0]);
    s[0][1] = wmma16(aQ[0][0], bk10, s[0][1]);
    s[1][1] = wmma16(aQ[1][0], bk10, s[1][1]);
    s[0][1] = wmma16(aQ[0][1], bk11, s[0][1]);
    s[1][1] = wmma16(aQ[1][1], bk11, s[1][1]);

    // ---- online softmax per M tile ----
    const int k0 = kt + ncol;
    const int k1 = kt + 16 + ncol;
#pragma unroll
    for (int mt = 0; mt < 2; ++mt) {
#pragma unroll
      for (int r = 0; r < 8; ++r) {
        const int tq = q0 + 16 * mt + rbase + r;
        float e0 = (k0 <= tq) ? s[mt][0][r] * scale : -1e30f;
        float e1 = (k1 <= tq) ? s[mt][1][r] * scale : -1e30f;
        float tm = fmaxf(e0, e1);
#pragma unroll
        for (int off = 1; off < 16; off <<= 1) tm = fmaxf(tm, __shfl_xor(tm, off));
        const float mnew  = fmaxf(mrow[mt][r], tm);
        const float alpha = __expf(mrow[mt][r] - mnew);
        const float p0 = __expf(e0 - mnew);
        const float p1 = __expf(e1 - mnew);
        float ps = p0 + p1;
#pragma unroll
        for (int off = 1; off < 16; off <<= 1) ps += __shfl_xor(ps, off);
        lrow[mt][r] = lrow[mt][r] * alpha + ps;
        mrow[mt][r] = mnew;
        o[mt][0][r] *= alpha; o[mt][1][r] *= alpha;
        o[mt][2][r] *= alpha; o[mt][3][r] *= alpha;
        const int row = 16 * mt + rbase + r;
        plds[wave][row * 32 + ncol]      = (_Float16)p0;
        plds[wave][row * 32 + ncol + 16] = (_Float16)p1;
      }
    }
    asm volatile("s_wait_dscnt 0" ::: "memory");  // wave-local C->A transpose

    // ---- O += P * V ----
    v16h aP0 = frag_a(pp);
    v16h aP1 = frag_a(pp + 16 * 32);
#pragma unroll
    for (int d = 0; d < 4; ++d) {
      v16h bv = frag_b(vp + (size_t)(d * 16) * T_ + kt);
      o[0][d] = wmma16(aP0, bv, o[0][d]);
      o[1][d] = wmma16(aP1, bv, o[1][d]);
    }
  }

  // ---- normalize + write [b][t][h*64+dh] ----
#pragma unroll
  for (int mt = 0; mt < 2; ++mt)
#pragma unroll
    for (int nt = 0; nt < 4; ++nt)
#pragma unroll
      for (int r = 0; r < 8; ++r) {
        const int t  = q0 + 16 * mt + rbase + r;
        const int dh = ncol + nt * 16;
        oh[((size_t)(b * T_ + t)) * D_ + h * DH_ + dh] =
            (_Float16)(o[mt][nt][r] / lrow[mt][r]);
      }
}

// -------------------------------------------------------------------------
// Kernel 3: output projection  out[b,t,d] = sum_o oh[b,t,o] * wo[d,o]
// One wave = 32x64 tile of the [4096 x 1024] result.
// -------------------------------------------------------------------------
__global__ void __launch_bounds__(256) oproj_kernel(
    const _Float16* __restrict__ oh, const _Float16* __restrict__ wo,
    float* __restrict__ out) {
  const int lane = threadIdx.x & 31;
  const int wave = threadIdx.x >> 5;
  const int tile = blockIdx.x * 8 + wave;
  const int m0 = (tile >> 4) * 32;
  const int d0 = (tile & 15) * 64;

  const _Float16* ap0 = oh + ((size_t)m0 + (lane & 15)) * D_ + ((lane >> 4) << 3);
  const _Float16* ap1 = ap0 + (size_t)16 * D_;
  const _Float16* bp0 = wo + ((size_t)d0 + (lane & 15)) * D_ + ((lane >> 4) << 3);
  const _Float16* bp1 = bp0 + (size_t)16 * D_;
  const _Float16* bp2 = bp0 + (size_t)32 * D_;
  const _Float16* bp3 = bp0 + (size_t)48 * D_;

  v8f acc[2][4] = {{v8f{}, v8f{}, v8f{}, v8f{}}, {v8f{}, v8f{}, v8f{}, v8f{}}};
#pragma unroll 4
  for (int kk = 0; kk < D_; kk += 32) {
    v16h a0 = frag_a(ap0 + kk);
    v16h a1 = frag_a(ap1 + kk);
    v16h b0 = frag_b(bp0 + kk);
    v16h b1 = frag_b(bp1 + kk);
    v16h b2 = frag_b(bp2 + kk);
    v16h b3 = frag_b(bp3 + kk);
    acc[0][0] = wmma16(a0, b0, acc[0][0]);
    acc[1][0] = wmma16(a1, b0, acc[1][0]);
    acc[0][1] = wmma16(a0, b1, acc[0][1]);
    acc[1][1] = wmma16(a1, b1, acc[1][1]);
    acc[0][2] = wmma16(a0, b2, acc[0][2]);
    acc[1][2] = wmma16(a1, b2, acc[1][2]);
    acc[0][3] = wmma16(a0, b3, acc[0][3]);
    acc[1][3] = wmma16(a1, b3, acc[1][3]);
  }
  const int rbase = (lane >> 4) << 3;
  const int ncol  = lane & 15;
#pragma unroll
  for (int mh = 0; mh < 2; ++mh)
#pragma unroll
    for (int nt = 0; nt < 4; ++nt)
#pragma unroll
      for (int r = 0; r < 8; ++r)
        out[(size_t)(m0 + 16 * mh + rbase + r) * D_ + d0 + nt * 16 + ncol] =
            acc[mh][nt][r];
}

// -------------------------------------------------------------------------
extern "C" void kernel_launch(void* const* d_in, const int* in_sizes, int n_in,
                              void* d_out, int out_size, void* d_ws, size_t ws_size,
                              hipStream_t stream) {
  // setup_inputs order: x, rope_freqs, mask, w_q, w_k, w_v, w_o (all fp32)
  const float* x    = (const float*)d_in[0];
  const float* rope = (const float*)d_in[1];
  // d_in[2] = mask: causal, computed analytically in-kernel.
  const float* wq = (const float*)d_in[3];
  const float* wk = (const float*)d_in[4];
  const float* wv = (const float*)d_in[5];
  const float* wo = (const float*)d_in[6];
  float* out = (float*)d_out;

  // workspace partition (f16 elements)
  _Float16* xh  = (_Float16*)d_ws;                      // [B*T][D]
  _Float16* wqh = xh  + (size_t)B_ * T_ * D_;           // [1024][1024]
  _Float16* wkh = wqh + (size_t)D_ * D_;                // [256][1024]
  _Float16* wvh = wkh + (size_t)256 * D_;               // [256][1024]
  _Float16* woh = wvh + (size_t)256 * D_;               // [1024][1024]
  _Float16* qh  = woh + (size_t)D_ * D_;                // [B][H][T][DH]
  _Float16* kh  = qh  + (size_t)B_ * H_ * T_ * DH_;     // [B][KV][T][DH]
  _Float16* vT  = kh  + (size_t)B_ * KV_ * T_ * DH_;    // [B][KV][DH][T]
  _Float16* oh  = vT  + (size_t)B_ * KV_ * T_ * DH_;    // [B][T][H*DH]

  const int nx  = B_ * T_ * D_ / 4;
  const int nwq = D_ * D_ / 4;
  const int nwk = 256 * D_ / 4;
  cvt_f32_f16_kernel<<<(nx  + 255) / 256, 256, 0, stream>>>(x,  xh,  nx);
  cvt_f32_f16_kernel<<<(nwq + 255) / 256, 256, 0, stream>>>(wq, wqh, nwq);
  cvt_f32_f16_kernel<<<(nwk + 255) / 256, 256, 0, stream>>>(wk, wkh, nwk);
  cvt_f32_f16_kernel<<<(nwk + 255) / 256, 256, 0, stream>>>(wv, wvh, nwk);
  cvt_f32_f16_kernel<<<(nwq + 255) / 256, 256, 0, stream>>>(wo, woh, nwq);

  // K1: (B*T/32) * 24 = 3072 waves
  qkv_rope_kernel<<<3072 / 8, 256, 0, stream>>>(xh, rope, wqh, wkh, wvh, qh, kh, vT);
  // K2: B*H*(T/32) = 2048 waves
  attn_kernel<<<2048 / 8, 256, 0, stream>>>(qh, kh, vT, oh);
  // K3: (B*T/32) * (D/64) = 2048 waves
  oproj_kernel<<<2048 / 8, 256, 0, stream>>>(oh, woh, out);
}
